// MultiHeadAttn1D_71279277244738
// MI455X (gfx1250) — compile-verified, matrix operand order checked by reading, not run on hardware
//
#include <hip/hip_runtime.h>
#include <hip/hip_bf16.h>

typedef __bf16 bf16_t;
typedef __attribute__((ext_vector_type(16))) __bf16 v16bf;
typedef __attribute__((ext_vector_type(8)))  float  v8f;
typedef __attribute__((ext_vector_type(4)))  int    v4i;

typedef __attribute__((address_space(1))) v4i gas_v4i;  // global int4
typedef __attribute__((address_space(3))) v4i las_v4i;  // LDS int4

#define TSEQ   2048
#define CIN    256
#define NHEADS 8
#define DHEAD  32
#define L2E    1.44269504088896340736f

__device__ __forceinline__ bf16_t f2bf(float f) { return (bf16_t)f; }

// 16-byte-per-lane async copy global -> LDS (ASYNCcnt-tracked).
__device__ __forceinline__ void async_cp16(void* lptr, const void* gptr) {
#if __has_builtin(__builtin_amdgcn_global_load_async_to_lds_b128)
    __builtin_amdgcn_global_load_async_to_lds_b128((gas_v4i*)gptr, (las_v4i*)lptr, 0, 0);
#else
    asm volatile("global_load_async_to_lds_b128 %0, %1, off"
                 :: "v"((unsigned)(unsigned long long)lptr), "v"(gptr) : "memory");
#endif
}
__device__ __forceinline__ void wait_async0() {
#if __has_builtin(__builtin_amdgcn_s_wait_asynccnt)
    __builtin_amdgcn_s_wait_asynccnt(0);
#else
    asm volatile("s_wait_asynccnt 0" ::: "memory");
#endif
}

// ---------------------------------------------------------------------------
// Kernel 0a: weights fp32 -> bf16, WB = [3][256][256] row-major.
// ---------------------------------------------------------------------------
__global__ __launch_bounds__(256)
void cvt_w_kernel(const float* __restrict__ Wq, const float* __restrict__ Wk,
                  const float* __restrict__ Wv, bf16_t* __restrict__ WB)
{
    int i   = blockIdx.x * 256 + threadIdx.x;   // 3*65536 total
    int sel = i >> 16, j = i & 0xFFFF;
    const float* W = (sel == 0) ? Wq : (sel == 1) ? Wk : Wv;
    WB[i] = f2bf(W[j]);
}

// ---------------------------------------------------------------------------
// Kernel 0b: x fp32 [n][c][t] -> bf16 transposed XT [n][t][c] via LDS tiles.
// ---------------------------------------------------------------------------
__global__ __launch_bounds__(256)
void cvt_transpose_x_kernel(const float* __restrict__ x, bf16_t* __restrict__ XT)
{
    __shared__ bf16_t tile[32][33];
    int b  = blockIdx.x;          // [n][ct][tt]
    int tt = b & 63;  b >>= 6;    // T/32 = 64
    int ct = b & 7;   b >>= 3;    // C/32 = 8
    int n  = b;
    int t0 = tt * 32, c0 = ct * 32;
    int tl = threadIdx.x & 31;
    int cl = threadIdx.x >> 5;    // 0..7
    const float* xb = x + (size_t)n * CIN * TSEQ;
#pragma unroll
    for (int r = 0; r < 4; ++r) {
        int c = cl + r * 8;
        tile[c][tl] = f2bf(xb[(size_t)(c0 + c) * TSEQ + t0 + tl]);
    }
    __syncthreads();
#pragma unroll
    for (int r = 0; r < 4; ++r) {
        int t = cl + r * 8;   // write coalesced along c
        XT[((size_t)n * TSEQ + t0 + t) * CIN + c0 + tl] = tile[tl][t];
    }
}

// ---------------------------------------------------------------------------
// Kernel 1: QKV projection. Wave computes 16(out) x 64(t) with K=256 loop.
// A (weights) loaded once per k-step, reused for 4 WMMAs. All operand loads
// are single 32B v16bf vector loads.
// Outputs: QT/KT [n][h][t][32] bf16, VT [n][h][d][T] bf16.
// ---------------------------------------------------------------------------
__global__ __launch_bounds__(256)
void qkv_proj_kernel(const bf16_t* __restrict__ XT, const bf16_t* __restrict__ WB,
                     bf16_t* __restrict__ QT, bf16_t* __restrict__ KT,
                     bf16_t* __restrict__ VT)
{
    const int lane = threadIdx.x & 31;
    const int wave = threadIdx.x >> 5;
    const int hf   = lane >> 4;
    const int l16  = lane & 15;

    int tile  = blockIdx.x * 8 + wave;   // 3*N*16*32 wave-tiles
    int tts   = tile & 31;  tile >>= 5;  // t super-tile (64 cols)
    int otile = tile & 15;  tile >>= 4;  // out-channel tile
    int n     = tile & 3;   tile >>= 2;
    int sel   = tile;                    // 0=Q 1=K 2=V

    const bf16_t* W  = WB + (size_t)sel * 65536;
    const bf16_t* xt = XT + (size_t)n * TSEQ * CIN;

    v8f acc[4] = {};
    for (int ks = 0; ks < 8; ++ks) {
        v16bf a = *(const v16bf*)(W + (size_t)(otile * 16 + l16) * CIN + ks * 32 + hf * 16);
#pragma unroll
        for (int u = 0; u < 4; ++u) {
            int t = tts * 64 + u * 16 + l16;
            v16bf b = *(const v16bf*)(xt + (size_t)t * CIN + ks * 32 + hf * 16);
            acc[u] = __builtin_amdgcn_wmma_f32_16x16x32_bf16(
                         false, a, false, b, (short)0, acc[u], false, false);
        }
    }

    // C layout: VGPR j, lane -> out = otile*16 + j + 8*hf, t = col l16
#pragma unroll
    for (int u = 0; u < 4; ++u) {
        int t = tts * 64 + u * 16 + l16;
        if (sel == 2) {
#pragma unroll
            for (int j = 0; j < 8; ++j) {
                int o = otile * 16 + j + hf * 8;
                VT[(size_t)(n * 256 + o) * TSEQ + t] = f2bf(acc[u][j]);
            }
        } else {
            bf16_t* dst = (sel == 0) ? QT : KT;
#pragma unroll
            for (int j = 0; j < 8; j += 2) {       // j,j+1 -> d,d+1 contiguous
                int o  = otile * 16 + j + hf * 8;  // even
                int hh = o >> 5, d = o & 31;
                union { bf16_t h[2]; unsigned u32; } pk;
                pk.h[0] = f2bf(acc[u][j]);
                pk.h[1] = f2bf(acc[u][j + 1]);
                *(unsigned*)(dst + (((size_t)(n * NHEADS + hh) * TSEQ + t) * DHEAD + d)) = pk.u32;
            }
        }
    }
}

// ---------------------------------------------------------------------------
// Kernel 2: flash attention. Block = 8 waves x 16 rows = 128 query rows.
// K/V 32x32 tiles staged in LDS with double-buffered async-to-LDS copies
// shared by all 8 waves; per 32-col step: 2 score WMMAs -> f32 online
// softmax -> bf16 P restripe through LDS -> 2 P*V WMMAs.
// ---------------------------------------------------------------------------
__global__ __launch_bounds__(256)
void attn_kernel(const bf16_t* __restrict__ QT, const bf16_t* __restrict__ KT,
                 const bf16_t* __restrict__ VT, float* __restrict__ out)
{
    __shared__ bf16_t kbuf[2][32][32];   // [s_local][d]
    __shared__ bf16_t vbuf[2][32][32];   // [d][s_local]
    __shared__ bf16_t pb[8][16][32];     // per-wave P tile

    const int tid  = threadIdx.x;
    const int lane = tid & 31;
    const int wave = tid >> 5;
    const int hf   = lane >> 4;
    const int l16  = lane & 15;

    int blk    = blockIdx.x;
    int rowBlk = blk & 15;               // T/128 = 16
    int nh     = blk >> 4;
    int n      = nh >> 3;
    int h      = nh & 7;
    int t0     = rowBlk * 128 + wave * 16;

    const bf16_t* qb = QT + (size_t)nh * TSEQ * DHEAD;
    const bf16_t* kb = KT + (size_t)nh * TSEQ * DHEAD;
    const bf16_t* vb = VT + (size_t)nh * DHEAD * TSEQ;

    v16bf aq = *(const v16bf*)(qb + (size_t)(t0 + l16) * DHEAD + hf * 16);

    v8f o0 = {}, o1 = {};
    float mrow[8], lrow[8];
#pragma unroll
    for (int j = 0; j < 8; ++j) { mrow[j] = -3.0e38f; lrow[j] = 0.0f; }

    const float scale = 0.17677669529663687f * L2E;
    bf16_t (*pw)[32] = pb[wave];

    // cooperative K/V tile fetch: 256 lanes x 16B = 4KB (K tile + V tile)
    auto issue = [&](int nb, int s0n) {
        if (tid < 128) {               // K tile: 2KB contiguous
            async_cp16(&kbuf[nb][0][0] + tid * 8,
                       kb + (size_t)s0n * DHEAD + tid * 8);
        } else {                       // V tile: 32 rows x 64B
            int i = tid - 128, d = i >> 2, part = i & 3;
            async_cp16(&vbuf[nb][0][0] + i * 8,
                       vb + (size_t)d * TSEQ + s0n + part * 8);
        }
    };

    issue(0, 0);
    wait_async0();
    __syncthreads();

    const int NITER = TSEQ / 32;
    for (int it = 0; it < NITER; ++it) {
        int cur = it & 1;
        if (it + 1 < NITER) issue(cur ^ 1, (it + 1) * 32);

        v16bf bk0 = *(const v16bf*)(&kbuf[cur][l16][hf * 16]);
        v16bf bk1 = *(const v16bf*)(&kbuf[cur][16 + l16][hf * 16]);

        v8f c0 = {}, c1 = {};
        c0 = __builtin_amdgcn_wmma_f32_16x16x32_bf16(false, aq, false, bk0, (short)0, c0, false, false);
        c1 = __builtin_amdgcn_wmma_f32_16x16x32_bf16(false, aq, false, bk1, (short)0, c1, false, false);

        float p0[8], p1[8], tm[8], ps[8];
#pragma unroll
        for (int j = 0; j < 8; ++j) {
            p0[j] = c0[j] * scale;
            p1[j] = c1[j] * scale;
            tm[j] = fmaxf(p0[j], p1[j]);
        }
#pragma unroll
        for (int m = 1; m < 16; m <<= 1) {
#pragma unroll
            for (int j = 0; j < 8; ++j) tm[j] = fmaxf(tm[j], __shfl_xor(tm[j], m, 32));
        }
        float fac[8];
#pragma unroll
        for (int j = 0; j < 8; ++j) {
            float mn = fmaxf(mrow[j], tm[j]);
            fac[j]  = exp2f(mrow[j] - mn);
            mrow[j] = mn;
            p0[j]   = exp2f(p0[j] - mn);
            p1[j]   = exp2f(p1[j] - mn);
            ps[j]   = p0[j] + p1[j];
        }
#pragma unroll
        for (int m = 1; m < 16; m <<= 1) {
#pragma unroll
            for (int j = 0; j < 8; ++j) ps[j] += __shfl_xor(ps[j], m, 32);
        }
#pragma unroll
        for (int j = 0; j < 8; ++j) {
            lrow[j] = lrow[j] * fac[j] + ps[j];
            o0[j] *= fac[j];
            o1[j] *= fac[j];
        }
        // restripe P (C-layout -> A-layout) through LDS as bf16
#pragma unroll
        for (int j = 0; j < 8; ++j) {
            int r = j + hf * 8;
            pw[r][l16]      = f2bf(p0[j]);
            pw[r][16 + l16] = f2bf(p1[j]);
        }
        asm volatile("s_wait_dscnt 0" ::: "memory");
        v16bf ap  = *(const v16bf*)(&pw[l16][hf * 16]);
        v16bf bv0 = *(const v16bf*)(&vbuf[cur][l16][hf * 16]);
        v16bf bv1 = *(const v16bf*)(&vbuf[cur][16 + l16][hf * 16]);

        o0 = __builtin_amdgcn_wmma_f32_16x16x32_bf16(false, ap, false, bv0, (short)0, o0, false, false);
        o1 = __builtin_amdgcn_wmma_f32_16x16x32_bf16(false, ap, false, bv1, (short)0, o1, false, false);

        wait_async0();     // next buffer resident
        __syncthreads();   // all waves done with cur before overwrite
    }

    // normalize + store: out[n][h*32+d][t], pack adjacent-t pairs into b64
#pragma unroll
    for (int j = 0; j < 8; j += 2) {
        int t = t0 + j + hf * 8;
        float iv0 = 1.0f / lrow[j], iv1 = 1.0f / lrow[j + 1];
        float2 s0 = make_float2(o0[j] * iv0, o0[j + 1] * iv1);
        float2 s1 = make_float2(o1[j] * iv0, o1[j + 1] * iv1);
        *(float2*)(out + (size_t)(n * 256 + h * DHEAD + l16)      * TSEQ + t) = s0;
        *(float2*)(out + (size_t)(n * 256 + h * DHEAD + 16 + l16) * TSEQ + t) = s1;
    }
}

// ---------------------------------------------------------------------------
extern "C" void kernel_launch(void* const* d_in, const int* in_sizes, int n_in,
                              void* d_out, int out_size, void* d_ws, size_t ws_size,
                              hipStream_t stream)
{
    const float* x  = (const float*)d_in[0];
    const float* Wq = (const float*)d_in[1];
    const float* Wk = (const float*)d_in[2];
    const float* Wv = (const float*)d_in[3];
    float* out = (float*)d_out;

    const int N = in_sizes[0] / (CIN * TSEQ);   // 4

    // Workspace: XT | WB | QT | KT | VT
    char* ws = (char*)d_ws;
    size_t xt_b  = (size_t)N * TSEQ * CIN * sizeof(bf16_t);
    size_t wb_b  = (size_t)3 * 256 * 256 * sizeof(bf16_t);
    wb_b = (wb_b + 255) & ~(size_t)255;
    size_t seg   = (size_t)N * NHEADS * TSEQ * DHEAD * sizeof(bf16_t);
    bf16_t* XT = (bf16_t*)ws;
    bf16_t* WB = (bf16_t*)(ws + xt_b);
    bf16_t* QT = (bf16_t*)(ws + xt_b + wb_b);
    bf16_t* KT = (bf16_t*)(ws + xt_b + wb_b + seg);
    bf16_t* VT = (bf16_t*)(ws + xt_b + wb_b + 2 * seg);

    cvt_w_kernel<<<(3 * 65536) / 256, 256, 0, stream>>>(Wq, Wk, Wv, WB);
    cvt_transpose_x_kernel<<<N * 8 * 64, 256, 0, stream>>>(x, XT);
    qkv_proj_kernel<<<(3 * N * 16 * 32) / 8, 256, 0, stream>>>(XT, WB, QT, KT, VT);
    attn_kernel<<<N * NHEADS * (TSEQ / 128), 256, 0, stream>>>(QT, KT, VT, out);
}
